// Model_3685081940550
// MI455X (gfx1250) — matrix operand order repaired and checked by reading, not compiled, not run on hardware
//
#include <hip/hip_runtime.h>
#include <hip/hip_bf16.h>
#include <math.h>

// ---------------------------------------------------------------------------
// Problem constants (B=16, N=512, DIN=128, DH=256)
// ---------------------------------------------------------------------------
#define BB   16
#define NN   512
#define DIN  128
#define DH   256
#define ROWS (BB * NN)            // 8192
#define NEGV (-9.0e15f)

typedef _Float16 half_t;
typedef __attribute__((ext_vector_type(4)))  float    f32x4;
typedef __attribute__((ext_vector_type(4)))  _Float16 h16x4;
typedef __attribute__((ext_vector_type(8)))  _Float16 v8h;
typedef __attribute__((ext_vector_type(16))) _Float16 v16h;
typedef __attribute__((ext_vector_type(8)))  float    v8f;

#define EPI_NONE       0
#define EPI_PRELU      1
#define EPI_BIAS_PRELU 2

// ---------------------------------------------------------------------------
// Block-tiled batched GEMM: C[b] = A[b](MxK) * B[b](KxN) (optionally B^T).
// Requirements: M%64==0, N%64==0, K%32==0.
// block = 128 threads (4 waves). Block tile: 64(M) x 64(N). Each wave owns a
// 16x64 strip: 4 accumulators, A fragment reused across 4 WMMAs per K-step.
// Double-buffered LDS tiles -> single barrier per K iteration.
// f32 -> b128 global loads -> f16 LDS -> v_wmma_f32_16x16x32_f16.
// ---------------------------------------------------------------------------
template<int TRANSB, int EPI>
__global__ __launch_bounds__(128)
void k_gemm_wmma(const float* __restrict__ A, const float* __restrict__ Bm,
                 float* __restrict__ C, int K,
                 long strA, long strB, long strC,
                 int lda, int ldb, int ldc,
                 const float* __restrict__ bias,
                 const float* __restrict__ alpha)
{
    A  += (long)blockIdx.z * strA;
    Bm += (long)blockIdx.z * strB;
    C  += (long)blockIdx.z * strC;

    const int tid  = threadIdx.x;        // 0..127
    const int lane = tid & 31;
    const int wave = tid >> 5;
    const int n0 = blockIdx.x * 64;
    const int m0 = blockIdx.y * 64;

    // A: 64 rows x 32 K; B: 64 cols x 32 K (K-contiguous per column).
    // Row stride 40 halves (80B): 16B-aligned for b128, staggers banks.
    // Double buffered: one barrier per K iteration.
    __shared__ __align__(16) half_t sA[2][64][40];
    __shared__ __align__(16) half_t sB[2][64][40];

    v8f acc[4] = {{}, {}, {}, {}};

    // fragment addressing (ISA 16-bit A 16x32 / B 32x16 layouts)
    const int arow    = wave * 16 + (lane & 15);
    const int half_lo = (lane & 16) ? 8 : 0;    // A: K sub-block
    const int kb      = (lane & 16) ? 8 : 0;    // B: K sub-block
    const int bcol    = lane & 15;

    // ---- precompute per-thread staging pointers (bump by stride per iter)
    const float* aptr[4];
    int ar_[4], akq_[4];
    #pragma unroll
    for (int i = 0; i < 4; ++i) {
        int idx = i * 128 + tid;                 // 512 f32x4 = 64 x 32 tile
        ar_[i] = idx >> 3; akq_[i] = idx & 7;
        aptr[i] = A + (long)(m0 + ar_[i]) * lda + akq_[i] * 4;
    }
    const float* bptr[4];
    int bi0_[4], bi1_[4];
    #pragma unroll
    for (int i = 0; i < 4; ++i) {
        int idx = i * 128 + tid;
        if (TRANSB) {                            // B^T: 64 rows x 32 K
            bi0_[i] = idx >> 3; bi1_[i] = idx & 7;   // c, kq
            bptr[i] = Bm + (long)(n0 + bi0_[i]) * ldb + bi1_[i] * 4;
        } else {                                 // B: 32 K-rows x 64 cols
            bi0_[i] = idx >> 4; bi1_[i] = idx & 15;  // kk, c4
            bptr[i] = Bm + (long)bi0_[i] * ldb + n0 + bi1_[i] * 4;
        }
    }
    const long bstep = TRANSB ? 32 : (long)32 * ldb;

    for (int k0 = 0; k0 < K; k0 += 32) {
        const int buf = (k0 >> 5) & 1;

        // ---- batch global loads into registers (b128, coalesced) ----
        f32x4 areg[4], breg[4];
        #pragma unroll
        for (int i = 0; i < 4; ++i) { areg[i] = *(const f32x4*)aptr[i]; aptr[i] += 32; }
        #pragma unroll
        for (int i = 0; i < 4; ++i) { breg[i] = *(const f32x4*)bptr[i]; bptr[i] += bstep; }
        if (k0 + 32 < K)  // prefetch next A K-tile (global_prefetch_b8)
            __builtin_prefetch(aptr[0], 0, 3);

        // ---- convert + store to LDS (ping-pong buffer) ----
        #pragma unroll
        for (int i = 0; i < 4; ++i) {
            h16x4 h = { (half_t)areg[i][0], (half_t)areg[i][1],
                        (half_t)areg[i][2], (half_t)areg[i][3] };
            *(h16x4*)&sA[buf][ar_[i]][akq_[i] * 4] = h;
        }
        #pragma unroll
        for (int i = 0; i < 4; ++i) {
            if (TRANSB) {
                h16x4 h = { (half_t)breg[i][0], (half_t)breg[i][1],
                            (half_t)breg[i][2], (half_t)breg[i][3] };
                *(h16x4*)&sB[buf][bi0_[i]][bi1_[i] * 4] = h;
            } else {
                sB[buf][bi1_[i] * 4 + 0][bi0_[i]] = (half_t)breg[i][0];
                sB[buf][bi1_[i] * 4 + 1][bi0_[i]] = (half_t)breg[i][1];
                sB[buf][bi1_[i] * 4 + 2][bi0_[i]] = (half_t)breg[i][2];
                sB[buf][bi1_[i] * 4 + 3][bi0_[i]] = (half_t)breg[i][3];
            }
        }
        __syncthreads();   // tiles staged (single barrier per iteration)

        // ---- A fragment: aligned 16B LDS loads ----
        v8h alo = *(const v8h*)&sA[buf][arow][half_lo];
        v8h ahi = *(const v8h*)&sA[buf][arow][16 + half_lo];
        v16h af;
        #pragma unroll
        for (int j = 0; j < 8; ++j) { af[j] = alo[j]; af[8 + j] = ahi[j]; }

        // ---- 4 WMMAs reusing the A fragment ----
        #pragma unroll
        for (int t = 0; t < 4; ++t) {
            v8h b0 = *(const v8h*)&sB[buf][t * 16 + bcol][kb];
            v8h b1 = *(const v8h*)&sB[buf][t * 16 + bcol][kb + 16];
            v16h bf;
            #pragma unroll
            for (int j = 0; j < 8; ++j) { bf[j] = b0[j]; bf[8 + j] = b1[j]; }
            acc[t] = __builtin_amdgcn_wmma_f32_16x16x32_f16(
                         false, af, false, bf, (short)0, acc[t], false, false);
        }
    }

    // ---- epilogue: VGPR r -> row r + (lane<16?0:8), col = lane&15 ----
    const int rbase = wave * 16 + ((lane & 16) ? 8 : 0);
    const float al = (EPI != EPI_NONE) ? *alpha : 0.0f;
    #pragma unroll
    for (int t = 0; t < 4; ++t) {
        const int col = n0 + t * 16 + bcol;
        const float bv = (EPI == EPI_BIAS_PRELU) ? bias[col] : 0.0f;
        #pragma unroll
        for (int r = 0; r < 8; ++r) {
            float x = acc[t][r];
            if (EPI == EPI_BIAS_PRELU) x += bv;
            if (EPI != EPI_NONE) x = (x >= 0.0f) ? x : al * x;
            C[(long)(m0 + rbase + r) * ldc + col] = x;
        }
    }
}

// ---------------------------------------------------------------------------
// Tiny single-wave GEMM (16x16 output) for graph_logits: C = A(16xK) * B^T.
// Same LDS layout as the big kernel: A rows / B^T rows K-contiguous with
// 80B row stride -> b128 staging loads, b64 LDS stores, b128 fragment loads.
// ---------------------------------------------------------------------------
__global__ __launch_bounds__(32)
void k_gemm_tiny(const float* __restrict__ A, const float* __restrict__ Bm,
                 float* __restrict__ C, int K, int lda, int ldb, int ldc)
{
    const int lane = threadIdx.x;
    __shared__ __align__(16) half_t ta[16][40];   // 16 rows x 32 K
    __shared__ __align__(16) half_t tb[16][40];   // 16 cols x 32 K (from B^T)
    v8f acc = {};
    const int arow    = lane & 15;
    const int half_lo = (lane & 16) ? 8 : 0;
    const int kb      = (lane & 16) ? 8 : 0;
    const int bcol    = lane & 15;

    // staging map: 128 f32x4 per 16x32 tile, 4 per lane
    const int sr_ = lane >> 3;          // row group base (0..3)
    const int skq_ = lane & 7;          // K-quad (0..7)

    for (int k0 = 0; k0 < K; k0 += 32) {
        f32x4 areg[4], breg[4];
        #pragma unroll
        for (int i = 0; i < 4; ++i) {
            int r = i * 4 + sr_;
            areg[i] = *(const f32x4*)(A  + (long)r * lda + (k0 + skq_ * 4));
            breg[i] = *(const f32x4*)(Bm + (long)r * ldb + (k0 + skq_ * 4));
        }
        #pragma unroll
        for (int i = 0; i < 4; ++i) {
            int r = i * 4 + sr_;
            h16x4 ha = { (half_t)areg[i][0], (half_t)areg[i][1],
                         (half_t)areg[i][2], (half_t)areg[i][3] };
            h16x4 hb = { (half_t)breg[i][0], (half_t)breg[i][1],
                         (half_t)breg[i][2], (half_t)breg[i][3] };
            *(h16x4*)&ta[r][skq_ * 4] = ha;
            *(h16x4*)&tb[r][skq_ * 4] = hb;
        }
        asm volatile("s_wait_dscnt 0" ::: "memory");  // single-wave RAW order

        v8h alo = *(const v8h*)&ta[arow][half_lo];
        v8h ahi = *(const v8h*)&ta[arow][16 + half_lo];
        v8h blo = *(const v8h*)&tb[bcol][kb];
        v8h bhi = *(const v8h*)&tb[bcol][kb + 16];
        v16h af, bf;
        #pragma unroll
        for (int j = 0; j < 8; ++j) {
            af[j] = alo[j]; af[8 + j] = ahi[j];
            bf[j] = blo[j]; bf[8 + j] = bhi[j];
        }
        acc = __builtin_amdgcn_wmma_f32_16x16x32_f16(
                  false, af, false, bf, (short)0, acc, false, false);
    }
    const int col = lane & 15;
    const int rbase = (lane & 16) ? 8 : 0;
    #pragma unroll
    for (int r = 0; r < 8; ++r)
        C[(long)(rbase + r) * ldc + col] = acc[r];
}

// ---------------------------------------------------------------------------
// Masked softmax + dropout multiply, in place on rows of [B*N, N]
// ---------------------------------------------------------------------------
__global__ __launch_bounds__(256)
void k_mask_softmax(float* __restrict__ sc, const float* __restrict__ adj,
                    const float* __restrict__ drop, int cols)
{
    const long row = blockIdx.x;
    float* s = sc + row * cols;
    const float* a = adj + row * cols;
    const float* d = drop + row * cols;
    __shared__ float red[256];
    const int t = threadIdx.x;

    float mx = -3.0e38f;
    for (int c = t; c < cols; c += 256) {
        float v = (a[c] > 0.0f) ? s[c] : NEGV;
        s[c] = v;
        mx = fmaxf(mx, v);
    }
    red[t] = mx; __syncthreads();
    for (int o = 128; o > 0; o >>= 1) {
        if (t < o) red[t] = fmaxf(red[t], red[t + o]);
        __syncthreads();
    }
    mx = red[0]; __syncthreads();

    float sum = 0.0f;
    for (int c = t; c < cols; c += 256) {
        float e = __expf(s[c] - mx);
        s[c] = e;
        sum += e;
    }
    red[t] = sum; __syncthreads();
    for (int o = 128; o > 0; o >>= 1) {
        if (t < o) red[t] += red[t + o];
        __syncthreads();
    }
    const float inv = 1.0f / red[0];
    for (int c = t; c < cols; c += 256) s[c] = s[c] * inv * d[c];
}

// ---------------------------------------------------------------------------
// Row-wise L2 normalize (D == 256 == blockDim.x), src -> dst (may alias)
// ---------------------------------------------------------------------------
__global__ __launch_bounds__(256)
void k_l2norm(const float* __restrict__ src, float* __restrict__ dst)
{
    const long row = blockIdx.x;
    const float* s = src + row * DH;
    float* o = dst + row * DH;
    __shared__ float red[256];
    const int t = threadIdx.x;
    const float x = s[t];
    red[t] = x * x; __syncthreads();
    for (int off = 128; off > 0; off >>= 1) {
        if (t < off) red[t] += red[t + off];
        __syncthreads();
    }
    const float nrm = fmaxf(sqrtf(red[0]), 1e-12f);
    o[t] = x / nrm;
}

// ---------------------------------------------------------------------------
// div_loss = mean over rows of ||q - k||^2   (atomic accumulation)
// ---------------------------------------------------------------------------
__global__ __launch_bounds__(256)
void k_divloss(const float* __restrict__ q, const float* __restrict__ k,
               float* __restrict__ dst)
{
    const long row = blockIdx.x;
    __shared__ float red[256];
    const int t = threadIdx.x;
    const float d = q[row * DH + t] - k[row * DH + t];
    red[t] = d * d; __syncthreads();
    for (int off = 128; off > 0; off >>= 1) {
        if (t < off) red[t] += red[t + off];
        __syncthreads();
    }
    if (t == 0) atomicAdd(dst, red[0] * (1.0f / (float)ROWS));
}

__global__ void k_zero(float* p) { p[0] = 0.0f; }

// ---------------------------------------------------------------------------
// Column mean over N: c[b,h] = mean_n h[b,n,h]
// ---------------------------------------------------------------------------
__global__ __launch_bounds__(256)
void k_mean(const float* __restrict__ h, float* __restrict__ c)
{
    const int b = blockIdx.x;
    const int t = threadIdx.x;                // h index, DH == 256
    const float* p = h + (long)b * NN * DH + t;
    float s = 0.0f;
    for (int n = 0; n < NN; ++n) s += p[(long)n * DH];
    c[b * DH + t] = s * (1.0f / (float)NN);
}

// ---------------------------------------------------------------------------
// Host-side launch sequence
// ---------------------------------------------------------------------------
static inline void launch_gemm(hipStream_t st,
                               const float* A, const float* B, float* C,
                               int M, int N, int K,
                               long sA, long sB, long sC,
                               int lda, int ldb, int ldc,
                               int batch, int transB, int epi,
                               const float* bias, const float* alpha)
{
    dim3 grid(N / 64, M / 64, batch);
    dim3 block(128, 1, 1);
    if (transB) {
        k_gemm_wmma<1, EPI_NONE><<<grid, block, 0, st>>>(
            A, B, C, K, sA, sB, sC, lda, ldb, ldc, bias, alpha);
    } else if (epi == EPI_PRELU) {
        k_gemm_wmma<0, EPI_PRELU><<<grid, block, 0, st>>>(
            A, B, C, K, sA, sB, sC, lda, ldb, ldc, bias, alpha);
    } else if (epi == EPI_BIAS_PRELU) {
        k_gemm_wmma<0, EPI_BIAS_PRELU><<<grid, block, 0, st>>>(
            A, B, C, K, sA, sB, sC, lda, ldb, ldc, bias, alpha);
    } else {
        k_gemm_wmma<0, EPI_NONE><<<grid, block, 0, st>>>(
            A, B, C, K, sA, sB, sC, lda, ldb, ldc, bias, alpha);
    }
}

extern "C" void kernel_launch(void* const* d_in, const int* in_sizes, int n_in,
                              void* d_out, int out_size, void* d_ws, size_t ws_size,
                              hipStream_t stream)
{
    const float* seq   = (const float*)d_in[0];
    const float* adj   = (const float*)d_in[1];
    const float* diff  = (const float*)d_in[2];
    const float* drop  = (const float*)d_in[3];
    const float* fc1_w = (const float*)d_in[4];
    const float* q_w   = (const float*)d_in[5];
    const float* k_w   = (const float*)d_in[6];
    const float* v_w1  = (const float*)d_in[7];
    const float* v_w2  = (const float*)d_in[8];
    const float* fc2_w = (const float*)d_in[9];
    const float* bias1 = (const float*)d_in[10];
    const float* bias2 = (const float*)d_in[11];
    const float* a1    = (const float*)d_in[12];   // prelu1_a
    const float* a2    = (const float*)d_in[13];   // prelu2_a
    const float* av    = (const float*)d_in[14];   // v_prelu_a

    float* outp = (float*)d_out;
    float* node_logits  = outp;                               // 8192*8192
    float* graph_logits = outp + (long)ROWS * ROWS;           // 16*16
    float* div_loss     = graph_logits + BB * BB;             // 1

    // workspace slabs (floats); reuse dead buffers
    const long SZ = (long)ROWS * DH;          // 2,097,152 floats
    float* ws  = (float*)d_ws;
    float* A0  = ws;                          // fts      -> agg  -> n1
    float* A1  = ws + SZ;                     // g2              -> n2
    float* A2  = ws + 2 * SZ;                 // out      -> t1
    float* A3  = ws + 3 * SZ;                 // q        -> h1
    float* A4  = ws + 4 * SZ;                 // k        -> h2
    float* SC  = ws + 5 * SZ;                 // scores/att: 16*512*512
    float* c1  = SC + (long)BB * NN * NN;
    float* c2  = c1 + BB * DH;

    const long sBMN = (long)NN * NN;          // 512*512 per-batch
    const long sBNH = (long)NN * DH;          // 512*256 per-batch

    // 1) fts = seq @ fc1_w        [8192,128]x[128,256]
    launch_gemm(stream, seq, fc1_w, A0, ROWS, DH, DIN, 0, 0, 0,
                DIN, DH, DH, 1, 0, EPI_NONE, nullptr, nullptr);
    // 2) g2 = seq @ fc2_w
    launch_gemm(stream, seq, fc2_w, A1, ROWS, DH, DIN, 0, 0, 0,
                DIN, DH, DH, 1, 0, EPI_NONE, nullptr, nullptr);
    // 3) out = adj @ fts   (batched)
    launch_gemm(stream, adj, A0, A2, NN, DH, NN, sBMN, sBNH, sBNH,
                NN, DH, DH, BB, 0, EPI_NONE, nullptr, nullptr);
    // 4) q_pre = out @ q_w ; 5) k_pre = out @ k_w
    launch_gemm(stream, A2, q_w, A3, ROWS, DH, DH, 0, 0, 0,
                DH, DH, DH, 1, 0, EPI_NONE, nullptr, nullptr);
    launch_gemm(stream, A2, k_w, A4, ROWS, DH, DH, 0, 0, 0,
                DH, DH, DH, 1, 0, EPI_NONE, nullptr, nullptr);
    // 6) l2-normalize q, k in place
    k_l2norm<<<dim3(ROWS), dim3(256), 0, stream>>>(A3, A3);
    k_l2norm<<<dim3(ROWS), dim3(256), 0, stream>>>(A4, A4);
    // 7) div_loss = mean ||q-k||^2
    k_zero<<<1, 1, 0, stream>>>(div_loss);
    k_divloss<<<dim3(ROWS), dim3(256), 0, stream>>>(A3, A4, div_loss);
    // 8) scores = q @ k^T  (batched, transB)
    launch_gemm(stream, A3, A4, SC, NN, NN, DH, sBNH, sBNH, sBMN,
                DH, DH, NN, BB, 1, EPI_NONE, nullptr, nullptr);
    // 9) att = softmax(mask(scores)) * drop_mask  (in place on SC)
    k_mask_softmax<<<dim3(ROWS), dim3(256), 0, stream>>>(SC, adj, drop, NN);
    // 10) agg = att @ out   (batched) -> A0 (fts dead)
    launch_gemm(stream, SC, A2, A0, NN, DH, NN, sBMN, sBNH, sBNH,
                NN, DH, DH, BB, 0, EPI_NONE, nullptr, nullptr);
    // 11) t1 = prelu(agg @ v_w1, v_a) -> A2 (out dead)
    launch_gemm(stream, A0, v_w1, A2, ROWS, DH, DH, 0, 0, 0,
                DH, DH, DH, 1, 0, EPI_PRELU, nullptr, av);
    // 12) h1 = prelu(t1 @ v_w2 + bias1, a1) -> A3 (q dead)
    launch_gemm(stream, A2, v_w2, A3, ROWS, DH, DH, 0, 0, 0,
                DH, DH, DH, 1, 0, EPI_BIAS_PRELU, bias1, a1);
    // 13) h2 = prelu(diff @ g2 + bias2, a2) (batched) -> A4 (k dead)
    launch_gemm(stream, diff, A1, A4, NN, DH, NN, sBMN, sBNH, sBNH,
                NN, DH, DH, BB, 0, EPI_BIAS_PRELU, bias2, a2);
    // 14) readout means
    k_mean<<<dim3(BB), dim3(256), 0, stream>>>(A3, c1);
    k_mean<<<dim3(BB), dim3(256), 0, stream>>>(A4, c2);
    // 15) n1 = l2n(h1) -> A0 ; n2 = l2n(h2) -> A1 ; normalize c1,c2 in place
    k_l2norm<<<dim3(ROWS), dim3(256), 0, stream>>>(A3, A0);
    k_l2norm<<<dim3(ROWS), dim3(256), 0, stream>>>(A4, A1);
    k_l2norm<<<dim3(BB), dim3(256), 0, stream>>>(c1, c1);
    k_l2norm<<<dim3(BB), dim3(256), 0, stream>>>(c2, c2);
    // 16) node_logits = n1 @ n2^T   [8192,256] x [8192,256]^T  (dominant)
    launch_gemm(stream, A0, A1, node_logits, ROWS, ROWS, DH, 0, 0, 0,
                DH, DH, ROWS, 1, 1, EPI_NONE, nullptr, nullptr);
    // 17) graph_logits = l2n(c1) @ l2n(c2)^T   [16,256] x [16,256]^T
    k_gemm_tiny<<<dim3(1), dim3(32), 0, stream>>>(c1, c2, graph_logits,
                                                  DH, DH, DH, BB);
}
